// LoopedGPT_9629316678117
// MI455X (gfx1250) — matrix operand order repaired and checked by reading, not compiled, not run on hardware
//
#include <hip/hip_runtime.h>
#include <math.h>

#define DMODEL 1024
#define NHEAD  16
#define HDIM   64
#define TSEQ   1024
#define BATCH  2
#define NTOK   (BATCH*TSEQ)

typedef __attribute__((ext_vector_type(8)))  unsigned short v8u;
typedef __attribute__((ext_vector_type(16))) unsigned short v16u;
typedef __attribute__((ext_vector_type(16))) __bf16        v16bf;
typedef __attribute__((ext_vector_type(8)))  float         v8f;

static __device__ inline unsigned short f32_to_bf16(float f) {
  unsigned u = __builtin_bit_cast(unsigned, f);
  u += 0x7FFFu + ((u >> 16) & 1u);          // round-to-nearest-even
  return (unsigned short)(u >> 16);
}

static __device__ inline v16bf make_frag(v8u lo, v8u hi) {
  v16u v = __builtin_shufflevector(lo, hi, 0,1,2,3,4,5,6,7,8,9,10,11,12,13,14,15);
  return __builtin_bit_cast(v16bf, v);
}

// A fragment (16xK=32 bf16 tile, row-major, row stride ld elements).
// wave32 layout: lane m=lane%16; K chunks at 8*half and 16+8*half.
static __device__ inline v16bf load_a_frag(const unsigned short* base, int ld, int lane) {
  int m = lane & 15, half = lane >> 4;
  const unsigned short* r = base + m * ld + half * 8;
  v8u a0 = *(const v8u*)(r);
  v8u a1 = *(const v8u*)(r + 16);
  return make_frag(a0, a1);
}

// B fragment (K=32 x 16) read from B^T layout [n][k], row stride ld elements.
// wave32 layout: lane n=lane%16; K chunks at 8*half and 16+8*half.
static __device__ inline v16bf load_b_frag(const unsigned short* base, int ld, int lane) {
  int n = lane & 15, half = lane >> 4;
  const unsigned short* r = base + n * ld + half * 8;
  v8u b0 = *(const v8u*)(r);
  v8u b1 = *(const v8u*)(r + 16);
  return make_frag(b0, b1);
}

static __device__ inline v8f wmma_bf16(v16bf a, v16bf b, v8f c) {
  return __builtin_amdgcn_wmma_f32_16x16x32_bf16(false, a, false, b, (short)0, c, false, false);
}

static __device__ inline v8f vzero8() {
  v8f v;
#pragma unroll
  for (int i = 0; i < 8; ++i) v[i] = 0.0f;
  return v;
}

static __device__ inline float redmax16(float v) {
#pragma unroll
  for (int m = 1; m < 16; m <<= 1) v = fmaxf(v, __shfl_xor(v, m, 32));
  return v;
}
static __device__ inline float redsum16(float v) {
#pragma unroll
  for (int m = 1; m < 16; m <<= 1) v += __shfl_xor(v, m, 32);
  return v;
}

static __device__ inline float gelu_t(float x) {
  const float c = 0.7978845608028654f;
  float u = c * (x + 0.044715f * x * x * x);
  return 0.5f * x * (1.0f + tanhf(u));
}

// ----------------------------------------------------------------------------
// fp32 -> bf16 conversion (weights, once per launch; n divisible by 256)
// ----------------------------------------------------------------------------
__global__ __launch_bounds__(256)
void cvt_bf16_k(const float* __restrict__ in, unsigned short* __restrict__ out, int n) {
  int i = blockIdx.x * 256 + threadIdx.x;
  if (i < n) out[i] = f32_to_bf16(in[i]);
}

// ----------------------------------------------------------------------------
// GEMM: C[M,N] = A[M,K] @ B[K,N] + bias (+ residual) (opt GELU)
// A,B bf16 in memory; block tile 128x128, BK=32, wave tile 32x64 (8 WMMA/step).
// out_bf16 selects bf16 or fp32 output.
// ----------------------------------------------------------------------------
__global__ __launch_bounds__(256)
void gemm_wmma_k(const unsigned short* __restrict__ A, const unsigned short* __restrict__ B,
                 const float* __restrict__ bias, const float* __restrict__ residual,
                 void* __restrict__ Cout, int M, int N, int K, int act, int out_bf16) {
  __shared__ __align__(32) unsigned short As[128 * 32];   // [m][k]
  __shared__ __align__(32) unsigned short Bs[128 * 32];   // B^T: [n][k]
  int tid = threadIdx.x, lane = tid & 31, wave = tid >> 5;
  int wm = (wave & 3) * 32;       // wave M offset within block tile
  int wn = (wave >> 2) * 64;      // wave N offset within block tile
  int mb = blockIdx.y * 128, nb = blockIdx.x * 128;

  v8f acc[2][4];
#pragma unroll
  for (int i = 0; i < 2; ++i)
#pragma unroll
    for (int j = 0; j < 4; ++j) acc[i][j] = vzero8();

  // staging assignments
  int ar = tid >> 1, ac = (tid & 1) * 16;   // A: 1 x v16u per thread
  int bk = tid >> 4, bn0 = (tid & 15) * 8;  // B: 2 x v8u per thread + LDS transpose

  for (int kb = 0; kb < K; kb += 32) {
    const unsigned short* ap = A + (size_t)(mb + ar) * K + kb + ac;
    *(v16u*)(As + ar * 32 + ac) = *(const v16u*)ap;
#pragma unroll
    for (int i = 0; i < 2; ++i) {
      int k = bk + i * 16;
      const unsigned short* bp = B + (size_t)(kb + k) * N + nb + bn0;
      v8u bv = *(const v8u*)bp;
#pragma unroll
      for (int j = 0; j < 8; ++j) Bs[(bn0 + j) * 32 + k] = bv[j];
    }
    if (kb + 32 < K) {           // prefetch next K tile (global_prefetch_b8)
      __builtin_prefetch(A + (size_t)(mb + ar) * K + kb + 32 + ac, 0, 1);
      __builtin_prefetch(B + (size_t)(kb + 32 + bk) * N + nb + bn0, 0, 1);
    }
    __syncthreads();
    v16bf af[2], bfb[4];
#pragma unroll
    for (int i = 0; i < 2; ++i) af[i] = load_a_frag(As + (wm + i * 16) * 32, 32, lane);
#pragma unroll
    for (int j = 0; j < 4; ++j) bfb[j] = load_b_frag(Bs + (wn + j * 16) * 32, 32, lane);
#pragma unroll
    for (int i = 0; i < 2; ++i)
#pragma unroll
      for (int j = 0; j < 4; ++j) acc[i][j] = wmma_bf16(af[i], bfb[j], acc[i][j]);
    __syncthreads();
  }

  int half = lane >> 4, lh = lane & 15;
#pragma unroll
  for (int i = 0; i < 2; ++i) {
#pragma unroll
    for (int j = 0; j < 4; ++j) {
#pragma unroll
      for (int e = 0; e < 8; ++e) {
        int gm = mb + wm + i * 16 + e + 8 * half;
        int gn = nb + wn + j * 16 + lh;
        float v = acc[i][j][e] + bias[gn];
        if (act) v = gelu_t(v);
        size_t o = (size_t)gm * N + gn;
        if (residual) v += residual[o];
        if (out_bf16) ((unsigned short*)Cout)[o] = f32_to_bf16(v);
        else          ((float*)Cout)[o] = v;
      }
    }
  }
}

// ----------------------------------------------------------------------------
// LayerNorm (optionally gathered from embedding table; fp32 or bf16 out).
// ----------------------------------------------------------------------------
__global__ __launch_bounds__(256)
void layernorm_k(const float* __restrict__ src, const int* __restrict__ idx,
                 const float* __restrict__ table, const float* __restrict__ g,
                 const float* __restrict__ b, void* __restrict__ out,
                 float eps, int out_bf16) {
  __shared__ float s1[256], s2[256];
  int row = blockIdx.x, tid = threadIdx.x;
  const float* r = idx ? (table + (size_t)idx[row] * DMODEL)
                       : (src + (size_t)row * DMODEL);
  float v[4]; float a = 0.f, q = 0.f;
#pragma unroll
  for (int i = 0; i < 4; ++i) { v[i] = r[tid + i * 256]; a += v[i]; q += v[i] * v[i]; }
  s1[tid] = a; s2[tid] = q; __syncthreads();
  for (int s = 128; s > 0; s >>= 1) {
    if (tid < s) { s1[tid] += s1[tid + s]; s2[tid] += s2[tid + s]; }
    __syncthreads();
  }
  float mean = s1[0] * (1.0f / DMODEL);
  float var  = s2[0] * (1.0f / DMODEL) - mean * mean;
  float rinv = rsqrtf(var + eps);
#pragma unroll
  for (int i = 0; i < 4; ++i) {
    int j = tid + i * 256;
    float o = (v[i] - mean) * rinv * g[j] + b[j];
    if (out_bf16) ((unsigned short*)out)[(size_t)row * DMODEL + j] = f32_to_bf16(o);
    else          ((float*)out)[(size_t)row * DMODEL + j] = o;
  }
}

__global__ __launch_bounds__(256)
void rmsnorm_k(const float* __restrict__ src, const float* __restrict__ wgt,
               unsigned short* __restrict__ out, float eps) {
  __shared__ float s2[256];
  int row = blockIdx.x, tid = threadIdx.x;
  const float* r = src + (size_t)row * DMODEL;
  float v[4]; float q = 0.f;
#pragma unroll
  for (int i = 0; i < 4; ++i) { v[i] = r[tid + i * 256]; q += v[i] * v[i]; }
  s2[tid] = q; __syncthreads();
  for (int s = 128; s > 0; s >>= 1) {
    if (tid < s) s2[tid] += s2[tid + s];
    __syncthreads();
  }
  float rinv = rsqrtf(s2[0] * (1.0f / DMODEL) + eps);
#pragma unroll
  for (int i = 0; i < 4; ++i) {
    int j = tid + i * 256;
    out[(size_t)row * DMODEL + j] = f32_to_bf16(v[i] * rinv * wgt[j]);
  }
}

// ----------------------------------------------------------------------------
// Split fused QKV (bf16) -> per-head Q,K [b,h,t,d] and V^T [b,h,d,t] (bf16)
// ----------------------------------------------------------------------------
__global__ __launch_bounds__(256)
void split_qkv_k(const unsigned short* __restrict__ qkv, unsigned short* __restrict__ q,
                 unsigned short* __restrict__ k, unsigned short* __restrict__ vt) {
  size_t i = (size_t)blockIdx.x * 256 + threadIdx.x;   // over NTOK*3072
  int c = (int)(i % (3 * DMODEL));
  int row = (int)(i / (3 * DMODEL));
  int b = row / TSEQ, t = row % TSEQ;
  int which = c >> 10, cc = c & 1023;
  int h = cc >> 6, d = cc & 63;
  unsigned short val = qkv[i];
  size_t bh = (size_t)b * NHEAD + h;
  if (which == 0)      q[(bh * TSEQ + t) * HDIM + d] = val;
  else if (which == 1) k[(bh * TSEQ + t) * HDIM + d] = val;
  else                 vt[(bh * HDIM + d) * TSEQ + t] = val;
}

// ----------------------------------------------------------------------------
// Flash-style causal attention with ALiBi.  One wave = 16 query rows.
// Grid: (T/128, NHEAD, BATCH), 256 threads (8 waves).  bf16 in / bf16 out.
// ----------------------------------------------------------------------------
__global__ __launch_bounds__(256)
void attn_k(const unsigned short* __restrict__ Q, const unsigned short* __restrict__ Kb,
            const unsigned short* __restrict__ Vt, unsigned short* __restrict__ Y) {
  __shared__ __align__(16) unsigned short P[8][16 * 32];   // per-wave P tile [m][k]
  int tid = threadIdx.x, lane = tid & 31, wave = tid >> 5;
  int hb = blockIdx.y, bb = blockIdx.z;
  int qw = blockIdx.x * 128 + wave * 16;
  size_t bh = (size_t)bb * NHEAD + hb;
  const unsigned short* Qh = Q  + bh * TSEQ * HDIM;
  const unsigned short* Kh = Kb + bh * TSEQ * HDIM;
  const unsigned short* Vh = Vt + bh * HDIM * TSEQ;
  int half = lane >> 4, lh = lane & 15;
  float slope = exp2f(-0.5f * (float)(hb + 1));
  const float scale = 0.125f;   // 1/sqrt(64)

  // Q fragments held in registers across the whole K loop
  v16bf aq0 = load_a_frag(Qh + (size_t)qw * HDIM,      HDIM, lane);  // d 0..31
  v16bf aq1 = load_a_frag(Qh + (size_t)qw * HDIM + 32, HDIM, lane);  // d 32..63

  v8f accv[4]; float rm[8], rs[8];
#pragma unroll
  for (int dt = 0; dt < 4; ++dt) accv[dt] = vzero8();
#pragma unroll
  for (int i = 0; i < 8; ++i) { rm[i] = -INFINITY; rs[i] = 0.0f; }

  int nch = qw / 32 + 1;     // 32-wide K chunks covering k <= qw+15
  for (int ch = 0; ch < nch; ++ch) {
    int k0 = ch * 32;
    v8f s[2];
#pragma unroll
    for (int j = 0; j < 2; ++j) {
      int kc = k0 + j * 16;
      v8f t = vzero8();
      t = wmma_bf16(aq0, load_b_frag(Kh + (size_t)kc * HDIM,      HDIM, lane), t);
      t = wmma_bf16(aq1, load_b_frag(Kh + (size_t)kc * HDIM + 32, HDIM, lane), t);
#pragma unroll
      for (int i = 0; i < 8; ++i) {
        int qi = qw + i + 8 * half;
        int ki = kc + lh;
        float bias = (qi >= ki) ? (-slope * (float)(qi - ki)) : -INFINITY;
        t[i] = t[i] * scale + bias;
      }
      s[j] = t;
    }
    // online softmax (row-wise across the 16 lanes of each half)
#pragma unroll
    for (int i = 0; i < 8; ++i) {
      float tm = redmax16(fmaxf(s[0][i], s[1][i]));
      float mn = fmaxf(rm[i], tm);
      float corr = __expf(rm[i] - mn);
      float p0 = __expf(s[0][i] - mn);
      float p1 = __expf(s[1][i] - mn);
      rs[i] = rs[i] * corr + redsum16(p0 + p1);
      rm[i] = mn;
#pragma unroll
      for (int dt = 0; dt < 4; ++dt) accv[dt][i] *= corr;
      int m = i + 8 * half;
      P[wave][m * 32 + lh]      = f32_to_bf16(p0);
      P[wave][m * 32 + 16 + lh] = f32_to_bf16(p1);
    }
    // P (C-layout) -> A-layout via wave-private LDS slab, then P @ V
    v16bf ap = load_a_frag(&P[wave][0], 32, lane);
#pragma unroll
    for (int dt = 0; dt < 4; ++dt)
      accv[dt] = wmma_bf16(ap, load_b_frag(Vh + (size_t)(dt * 16) * TSEQ + k0, TSEQ, lane), accv[dt]);
  }

  // epilogue: normalize and scatter back to [b, t, h*64+d] as bf16
#pragma unroll
  for (int i = 0; i < 8; ++i) {
    int qi = qw + i + 8 * half;
    float inv = 1.0f / rs[i];
#pragma unroll
    for (int dt = 0; dt < 4; ++dt) {
      int d = dt * 16 + lh;
      Y[((size_t)(bb * TSEQ + qi) * NHEAD + hb) * HDIM + d] = f32_to_bf16(accv[dt][i] * inv);
    }
  }
}

// ----------------------------------------------------------------------------
extern "C" void kernel_launch(void* const* d_in, const int* in_sizes, int n_in,
                              void* d_out, int out_size, void* d_ws, size_t ws_size,
                              hipStream_t stream) {
  (void)in_sizes; (void)n_in; (void)out_size; (void)ws_size;
  const int*   idx    = (const int*)d_in[0];
  const float* tok    = (const float*)d_in[1];
  const float* ln_e_g = (const float*)d_in[2];
  const float* ln_e_b = (const float*)d_in[3];
  const float* Wqkv   = (const float*)d_in[4];
  const float* bqkv   = (const float*)d_in[5];
  const float* Wo     = (const float*)d_in[6];
  const float* bo     = (const float*)d_in[7];
  const float* W1     = (const float*)d_in[8];
  const float* b1     = (const float*)d_in[9];
  const float* W2     = (const float*)d_in[10];
  const float* b2     = (const float*)d_in[11];
  const float* n1w    = (const float*)d_in[12];
  const float* n2w    = (const float*)d_in[13];
  const float* lnfg   = (const float*)d_in[14];
  const float* lnfb   = (const float*)d_in[15];
  const float* Wlm    = (const float*)d_in[16];
  const float* blm    = (const float*)d_in[17];
  float* out = (float*)d_out;

  char* w = (char*)d_ws;
  auto take = [&](size_t bytes) { char* p = w; w += (bytes + 255) & ~(size_t)255; return p; };
  // bf16 weight cache (re-built every launch; deterministic)
  unsigned short* wqkv_b = (unsigned short*)take((size_t)DMODEL * 3 * DMODEL * 2);
  unsigned short* wo_b   = (unsigned short*)take((size_t)DMODEL * DMODEL * 2);
  unsigned short* w1_b   = (unsigned short*)take((size_t)DMODEL * 4 * DMODEL * 2);
  unsigned short* w2_b   = (unsigned short*)take((size_t)4 * DMODEL * DMODEL * 2);
  unsigned short* wlm_b  = (unsigned short*)take((size_t)DMODEL * 32000 * 2);
  // activations
  float*          x    = (float*)take((size_t)NTOK * DMODEL * 4);          // residual (fp32)
  unsigned short* hb   = (unsigned short*)take((size_t)NTOK * DMODEL * 2); // normed (bf16)
  unsigned short* qkvb = (unsigned short*)take((size_t)NTOK * 3 * DMODEL * 2);
  unsigned short* yb   = (unsigned short*)take((size_t)NTOK * DMODEL * 2);
  unsigned short* gb   = (unsigned short*)take((size_t)NTOK * 4 * DMODEL * 2);
  unsigned short* qb   = (unsigned short*)take((size_t)NTOK * DMODEL * 2);
  unsigned short* kb   = (unsigned short*)take((size_t)NTOK * DMODEL * 2);
  unsigned short* vt   = (unsigned short*)take((size_t)NTOK * DMODEL * 2);

  dim3 blk(256);
  const float EPS_LN  = 1e-5f;
  const float EPS_RMS = 1.1920929e-7f;

  // weights -> bf16
  cvt_bf16_k<<<(DMODEL*3*DMODEL)/256, blk, 0, stream>>>(Wqkv, wqkv_b, DMODEL*3*DMODEL);
  cvt_bf16_k<<<(DMODEL*DMODEL)/256,   blk, 0, stream>>>(Wo,   wo_b,   DMODEL*DMODEL);
  cvt_bf16_k<<<(DMODEL*4*DMODEL)/256, blk, 0, stream>>>(W1,   w1_b,   DMODEL*4*DMODEL);
  cvt_bf16_k<<<(4*DMODEL*DMODEL)/256, blk, 0, stream>>>(W2,   w2_b,   4*DMODEL*DMODEL);
  cvt_bf16_k<<<(DMODEL*32000)/256,    blk, 0, stream>>>(Wlm,  wlm_b,  DMODEL*32000);

  // x = layernorm(tok_embed[idx])  (fp32 residual stream)
  layernorm_k<<<NTOK, blk, 0, stream>>>(nullptr, idx, tok, ln_e_g, ln_e_b, x, EPS_LN, 0);

  for (int L = 0; L < 4; ++L) {
    rmsnorm_k<<<NTOK, blk, 0, stream>>>(x, n1w, hb, EPS_RMS);
    gemm_wmma_k<<<dim3(3 * DMODEL / 128, NTOK / 128), blk, 0, stream>>>(
        hb, wqkv_b, bqkv, nullptr, qkvb, NTOK, 3 * DMODEL, DMODEL, 0, 1);
    split_qkv_k<<<(NTOK * 3 * DMODEL) / 256, blk, 0, stream>>>(qkvb, qb, kb, vt);
    attn_k<<<dim3(TSEQ / 128, NHEAD, BATCH), blk, 0, stream>>>(qb, kb, vt, yb);
    gemm_wmma_k<<<dim3(DMODEL / 128, NTOK / 128), blk, 0, stream>>>(
        yb, wo_b, bo, x, x, NTOK, DMODEL, DMODEL, 0, 0);
    rmsnorm_k<<<NTOK, blk, 0, stream>>>(x, n2w, hb, EPS_RMS);
    gemm_wmma_k<<<dim3(4 * DMODEL / 128, NTOK / 128), blk, 0, stream>>>(
        hb, w1_b, b1, nullptr, gb, NTOK, 4 * DMODEL, DMODEL, 1, 1);
    gemm_wmma_k<<<dim3(DMODEL / 128, NTOK / 128), blk, 0, stream>>>(
        gb, w2_b, b2, x, x, NTOK, DMODEL, 4 * DMODEL, 0, 0);
  }

  layernorm_k<<<NTOK, blk, 0, stream>>>(x, nullptr, nullptr, lnfg, lnfb, hb, EPS_LN, 1);
  gemm_wmma_k<<<dim3(32000 / 128, NTOK / 128), blk, 0, stream>>>(
      hb, wlm_b, blm, nullptr, out, NTOK, 32000, DMODEL, 0, 0);
}
